// ParallelEuclideanEncoder_41300405518820
// MI455X (gfx1250) — matrix operand order repaired and checked by reading, not compiled, not run on hardware
//
#include <hip/hip_runtime.h>

#define S_     5
#define BB     16
#define LL     4096
#define DM     64
#define DI     128
#define NPAD   48
#define EMB    32

typedef __attribute__((ext_vector_type(16))) _Float16 v16h;
typedef __attribute__((ext_vector_type(8)))  _Float16 v8h;
typedef __attribute__((ext_vector_type(2)))  _Float16 v2h;
typedef __attribute__((ext_vector_type(8)))  float    v8f;

union AFrag { v16h v; v8h h8[2]; };

__device__ __forceinline__ float sigmoidf_(float x){ return 1.f/(1.f+__expf(-x)); }
__device__ __forceinline__ float siluf_(float x){ return x*sigmoidf_(x); }

// A fragment (16x32 f16, row-major in memory): lane<16 holds K[kb..kb+7],K[kb+16..kb+23]
__device__ __forceinline__ v16h ldA(const _Float16* row64, int kbase, int lane){
  AFrag f; int koff = kbase + ((lane>>4)<<3);
  f.h8[0] = *(const v8h*)(row64 + koff);
  f.h8[1] = *(const v8h*)(row64 + koff + 16);
  return f.v;
}
// B fragment from LDS W^T[N][Ktot]: lane<16 holds K[kb..kb+7],K[kb+16..kb+23]; lanes 16-31 K[kb+8..kb+15],K[kb+24..kb+31]
__device__ __forceinline__ v16h ldB(const _Float16* wT, int Ktot, int kbase, int nAbs, int lane){
  AFrag f; int k = kbase + ((lane>>4)<<3);
  const _Float16* p = wT + nAbs*Ktot + k;
  f.h8[0] = *(const v8h*)(p);
  f.h8[1] = *(const v8h*)(p+16);
  return f.v;
}
#define WMMA(a,b,c) __builtin_amdgcn_wmma_f32_16x16x32_f16(false,(a),false,(b),(short)0,(c),false,false)

// ---------------- input embedding: h = x*in_W + in_b ----------------
__global__ __launch_bounds__(256) void k_inembed(
    const float* s0,const float* s1,const float* s2,const float* s3,const float* s4,
    const float* inW,const float* inb, float* h){
  long idx = (long)blockIdx.x*256 + threadIdx.x;
  int d = (int)(idx & 63); long row = idx >> 6;
  int s = (int)(row / (BB*(long)LL));
  long lrow = row - (long)s*BB*LL;
  const float* xs = (s==0)?s0:(s==1)?s1:(s==2)?s2:(s==3)?s3:s4;
  h[idx] = xs[lrow]*inW[s*DM+d] + inb[s*DM+d];
}

// ---------------- rmsnorm -> f16 ----------------
__global__ __launch_bounds__(256) void k_rmsnorm(const float* h, const float* nw, _Float16* xn16){
  int lane = threadIdx.x & 31;
  long row = ((long)blockIdx.x*256 + threadIdx.x) >> 5;
  int s = (int)(row / (BB*(long)LL));
  float2 v = ((const float2*)(h + row*DM))[lane];
  float ss = v.x*v.x + v.y*v.y;
  ss += __shfl_xor(ss,1,32); ss += __shfl_xor(ss,2,32);
  ss += __shfl_xor(ss,4,32); ss += __shfl_xor(ss,8,32); ss += __shfl_xor(ss,16,32);
  float sc = rsqrtf(ss*(1.f/DM) + 1e-5f);
  float2 w = ((const float2*)(nw + s*DM))[lane];
  v2h o; o.x = (_Float16)(v.x*sc*w.x); o.y = (_Float16)(v.y*sc*w.y);
  ((v2h*)(xn16 + row*DM))[lane] = o;
}

// ---------------- in_proj GEMM: [rows,64]x[64,256] -> xi(128) | silu(res)(128) ----------------
__global__ __launch_bounds__(256) void k_inproj(const _Float16* xn16, const float* Wg,
                                                float* xi, float* res_silu){
  __shared__ __align__(16) _Float16 wT[2*DI*DM]; // [256][64] f16, 32KB
  int blk = blockIdx.x;
  const int tilesPer = LL/64;
  int s = blk / (BB*tilesPer);
  int rem = blk % (BB*tilesPer);
  int b = rem / tilesPer, tc = rem % tilesPer;
  const float* W = Wg + s*DM*2*DI;
  for (int i = threadIdx.x; i < DM*2*DI; i += 256){
    int k = i >> 8, n = i & 255;
    wT[n*DM + k] = (_Float16)W[i];
  }
  __syncthreads();
  int wave = threadIdx.x >> 5, lane = threadIdx.x & 31;
  int mt = wave >> 1, nh = wave & 1;
  long row0 = ((long)(s*BB + b)*LL + tc*64 + mt*16);
  const _Float16* arow = xn16 + (row0 + (lane & 15))*DM;
  v8f zz = {0,0,0,0,0,0,0,0};
  v8f acc[8];
  #pragma unroll
  for (int i=0;i<8;++i) acc[i]=zz;
  #pragma unroll
  for (int c=0;c<2;++c){
    v16h a = ldA(arow, c*32, lane);
    #pragma unroll
    for (int nt=0;nt<8;++nt){
      v16h bm = ldB(wT, DM, c*32, (nh*8+nt)*16 + (lane&15), lane);
      acc[nt] = WMMA(a, bm, acc[nt]);
    }
  }
  int nb = lane & 15, ro = (lane>>4)*8;
  #pragma unroll
  for (int nt=0;nt<8;++nt){
    int col = (nh*8+nt)*16 + nb;
    #pragma unroll
    for (int j=0;j<8;++j){
      long r = row0 + ro + j;
      float v = acc[nt][j];
      if (nh==0) xi[r*DI + col] = v;
      else       res_silu[r*DI + (col-128)] = siluf_(v);
    }
  }
}

// ---------------- causal depthwise conv (D_CONV=4) + SiLU ----------------
__global__ __launch_bounds__(256) void k_conv(const float* xi, const float* cw, const float* cb,
                                              float* u, _Float16* u16){
  long idx = (long)blockIdx.x*256 + threadIdx.x;
  int d = (int)(idx & 127);
  long r = idx >> 7;
  int chunk = (int)(r & 63);
  int sb = (int)(r >> 6);
  int s = sb >> 4;
  const float* w = cw + (s*DI + d)*4;
  float w0=w[0],w1=w[1],w2=w[2],w3=w[3];
  float bias = cb[s*DI + d];
  long base = (long)sb*LL;
  int t0 = chunk*64;
  float x1=0.f,x2=0.f,x3=0.f;
  if (t0 >= 1) x1 = xi[(base + t0-1)*DI + d];
  if (t0 >= 2) x2 = xi[(base + t0-2)*DI + d];
  if (t0 >= 3) x3 = xi[(base + t0-3)*DI + d];
  for (int t=t0; t<t0+64; ++t){
    float x0 = xi[(base + t)*DI + d];
    float v = siluf_(w0*x3 + w1*x2 + w2*x1 + w3*x0 + bias);
    long o = (base + t)*DI + d;
    u[o] = v;
    u16[o] = (_Float16)v;
    x3=x2; x2=x1; x1=x0;
  }
}

// ---------------- x_proj GEMM: [rows,128]x[128,36(pad48)] ----------------
__global__ __launch_bounds__(256) void k_xproj(const _Float16* u16, const float* Wg, float* xdbl){
  __shared__ __align__(16) _Float16 wT[NPAD*DI]; // 12KB
  int blk = blockIdx.x;
  const int tilesPer = LL/128;
  int s = blk / (BB*tilesPer);
  int rem = blk % (BB*tilesPer);
  int b = rem / tilesPer, tc = rem % tilesPer;
  const float* W = Wg + s*DI*36;
  for (int i=threadIdx.x;i<NPAD*DI;i+=256){
    int n = i / DI, k = i % DI;
    wT[n*DI + k] = (n < 36) ? (_Float16)W[k*36 + n] : (_Float16)0.f;
  }
  __syncthreads();
  int wave = threadIdx.x>>5, lane = threadIdx.x&31;
  long row0 = ((long)(s*BB+b)*LL + tc*128 + wave*16);
  const _Float16* arow = u16 + (row0 + (lane&15))*DI;
  v8f zz={0,0,0,0,0,0,0,0};
  v8f acc[3]; acc[0]=zz;acc[1]=zz;acc[2]=zz;
  #pragma unroll
  for (int c=0;c<4;++c){
    v16h a = ldA(arow, c*32, lane);
    #pragma unroll
    for (int nt=0;nt<3;++nt){
      v16h bm = ldB(wT, DI, c*32, nt*16 + (lane&15), lane);
      acc[nt] = WMMA(a,bm,acc[nt]);
    }
  }
  int nb = lane&15, ro = (lane>>4)*8;
  #pragma unroll
  for (int nt=0;nt<3;++nt){
    #pragma unroll
    for (int j=0;j<8;++j){
      long rr = row0 + ro + j;
      xdbl[rr*NPAD + nt*16 + nb] = acc[nt][j];
    }
  }
}

// ---------------- delta = softplus(dt @ dt_W + dt_b) ----------------
__global__ __launch_bounds__(256) void k_delta(const float* xdbl, const float* dtW,
                                               const float* dtb, float* delta){
  long idx = (long)blockIdx.x*256 + threadIdx.x;
  int d = (int)(idx & 127);
  long row = idx >> 7;
  int s = (int)(row / (BB*(long)LL));
  const float* xb = xdbl + row*NPAD;
  const float* Wd = dtW + s*4*DI + d;
  float x = dtb[s*DI + d];
  x += xb[0]*Wd[0] + xb[1]*Wd[DI] + xb[2]*Wd[2*DI] + xb[3]*Wd[3*DI];
  delta[row*DI + d] = (x > 20.f) ? x : log1pf(__expf(x));
}

// ---------------- selective scan: wave = 2 channels x 16 states ----------------
__global__ __launch_bounds__(256) void k_scan(const float* delta, const float* u, const float* xdbl,
                                              const float* res_silu, const float* Alog, const float* Dp,
                                              _Float16* y16){
  int gw = (int)(((long)blockIdx.x*256 + threadIdx.x) >> 5);
  int lane = threadIdx.x & 31;
  int dpair = gw & 63; int sb = gw >> 6;           // sb in [0,80)
  int s = sb >> 4;
  int d = dpair*2 + (lane >> 4);
  int n = lane & 15;
  float An = -__expf(Alog[(s*DI + d)*16 + n]);
  float Dd = Dp[s*DI + d];
  long rowbase = (long)sb * LL;
  float hst = 0.f;
  for (int t = 0; t < LL; ++t){
    long row = rowbase + t;
    if ((t & 7) == 0 && t + 16 < LL){
      __builtin_prefetch(delta + (row+16)*DI + d, 0, 1);   // global_prefetch_b8
      __builtin_prefetch(xdbl  + (row+16)*NPAD,   0, 1);
    }
    float dl = delta[row*DI + d];
    float uu = u[row*DI + d];
    float Bn = xdbl[row*NPAD + 4 + n];
    float Cn = xdbl[row*NPAD + 20 + n];
    float dA = __expf(dl * An);
    hst = hst * dA + (dl * uu) * Bn;
    float r = hst * Cn;
    r += __shfl_xor(r, 1, 32);
    r += __shfl_xor(r, 2, 32);
    r += __shfl_xor(r, 4, 32);
    r += __shfl_xor(r, 8, 32);
    if (n == 0){
      float y = (r + uu * Dd) * res_silu[row*DI + d];
      y16[row*DI + d] = (_Float16)y;
    }
  }
}

// ---------------- out_proj GEMM + residual add into h ----------------
__global__ __launch_bounds__(256) void k_outproj(const _Float16* y16, const float* Wg, float* h){
  __shared__ __align__(16) _Float16 wT[DM*DI]; // [64][128] 16KB
  int blk = blockIdx.x;
  const int tilesPer = LL/128;
  int s = blk / (BB*tilesPer);
  int rem = blk % (BB*tilesPer);
  int b = rem / tilesPer, tc = rem % tilesPer;
  const float* W = Wg + s*DI*DM;
  for (int i=threadIdx.x;i<DM*DI;i+=256){
    int n = i >> 7, k = i & 127;
    wT[n*DI + k] = (_Float16)W[k*DM + n];
  }
  __syncthreads();
  int wave = threadIdx.x>>5, lane = threadIdx.x&31;
  long row0 = ((long)(s*BB+b)*LL + tc*128 + wave*16);
  const _Float16* arow = y16 + (row0 + (lane&15))*DI;
  v8f zz={0,0,0,0,0,0,0,0};
  v8f acc[4];
  #pragma unroll
  for (int i=0;i<4;++i) acc[i]=zz;
  #pragma unroll
  for (int c=0;c<4;++c){
    v16h a = ldA(arow, c*32, lane);
    #pragma unroll
    for (int nt=0;nt<4;++nt){
      v16h bm = ldB(wT, DI, c*32, nt*16 + (lane&15), lane);
      acc[nt] = WMMA(a,bm,acc[nt]);
    }
  }
  int nb = lane&15, ro = (lane>>4)*8;
  #pragma unroll
  for (int nt=0;nt<4;++nt){
    #pragma unroll
    for (int j=0;j<8;++j){
      long rr = row0 + ro + j;
      h[rr*DM + nt*16 + nb] += acc[nt][j];
    }
  }
}

// ---------------- final rmsnorm + mean over L ----------------
__global__ __launch_bounds__(256) void k_finalmean(const float* h, const float* nw, float* hm){
  __shared__ float part[8][DM];
  int sb = blockIdx.x; int s = sb >> 4;
  int wave = threadIdx.x >> 5, lane = threadIdx.x & 31;
  float2 w = ((const float2*)(nw + s*DM))[lane];
  float a0 = 0.f, a1 = 0.f;
  for (int t = wave; t < LL; t += 8){
    long row = (long)sb*LL + t;
    float2 v = ((const float2*)(h + row*DM))[lane];
    float ss = v.x*v.x + v.y*v.y;
    ss += __shfl_xor(ss,1,32); ss += __shfl_xor(ss,2,32);
    ss += __shfl_xor(ss,4,32); ss += __shfl_xor(ss,8,32); ss += __shfl_xor(ss,16,32);
    float sc = rsqrtf(ss*(1.f/DM) + 1e-5f);
    a0 += v.x*sc*w.x; a1 += v.y*sc*w.y;
  }
  part[wave][lane*2]   = a0;
  part[wave][lane*2+1] = a1;
  __syncthreads();
  if (wave == 0){
    float s0=0.f, s1=0.f;
    #pragma unroll
    for (int i=0;i<8;++i){ s0 += part[i][lane*2]; s1 += part[i][lane*2+1]; }
    hm[sb*DM + lane*2]   = s0*(1.f/LL);
    hm[sb*DM + lane*2+1] = s1*(1.f/LL);
  }
}

// ---------------- head: z = hm @ out_W + out_b (rows 0..4 of output) ----------------
__global__ __launch_bounds__(256) void k_head(const float* hm, const float* outW,
                                              const float* outb, float* out){
  int idx = blockIdx.x*256 + threadIdx.x;
  if (idx >= S_*BB*EMB) return;
  int e = idx & 31; int t = idx >> 5; int b = t & 15; int s = t >> 4;
  float acc = outb[s*EMB + e];
  const float* hv = hm + (s*BB + b)*DM;
  #pragma unroll 8
  for (int k=0;k<DM;++k) acc += hv[k]*outW[(s*DM + k)*EMB + e];
  out[(s*BB + b)*EMB + e] = acc;
}

// ---------------- combined = sum over branches (row 5) ----------------
__global__ __launch_bounds__(256) void k_combine(float* out){
  int idx = blockIdx.x*256 + threadIdx.x;
  if (idx >= BB*EMB) return;
  float acc = 0.f;
  #pragma unroll
  for (int s=0;s<S_;++s) acc += out[s*BB*EMB + idx];
  out[S_*BB*EMB + idx] = acc;
}

extern "C" void kernel_launch(void* const* d_in, const int* in_sizes, int n_in,
                              void* d_out, int out_size, void* d_ws, size_t ws_size,
                              hipStream_t stream){
  (void)in_sizes; (void)n_in; (void)out_size; (void)ws_size;
  const float* sig[5]; for (int i=0;i<5;++i) sig[i] = (const float*)d_in[i];
  const float* inW   = (const float*)d_in[5];
  const float* inb   = (const float*)d_in[6];
  const float* fnorm = (const float*)d_in[27];
  const float* outW  = (const float*)d_in[28];
  const float* outb  = (const float*)d_in[29];

  char* ws = (char*)d_ws;
  const long ROWS = (long)S_*BB*LL;                 // 327,680
  size_t off = 0;
  float*    h    = (float*)(ws+off);    off += (size_t)ROWS*DM*4;
  float*    xi   = (float*)(ws+off);    off += (size_t)ROWS*DI*4;   // aliased by delta
  float*    resb = (float*)(ws+off);    off += (size_t)ROWS*DI*4;
  float*    u    = (float*)(ws+off);    off += (size_t)ROWS*DI*4;
  _Float16* xn16 = (_Float16*)(ws+off); off += (size_t)ROWS*DM*2;
  _Float16* u16  = (_Float16*)(ws+off); off += (size_t)ROWS*DI*2;   // aliased by y16
  float*    xdbl = (float*)(ws+off);    off += (size_t)ROWS*NPAD*4;
  float*    hm   = (float*)(ws+off);    off += (size_t)S_*BB*DM*4;
  float*    delta = xi;
  _Float16* y16  = u16;
  float*    out  = (float*)d_out;

  k_inembed<<<(unsigned)(ROWS*DM/256),256,0,stream>>>(sig[0],sig[1],sig[2],sig[3],sig[4],inW,inb,h);

  for (int l=0;l<2;++l){
    int pb = 7 + l*10;
    const float* normw  = (const float*)d_in[pb+0];
    const float* inproj = (const float*)d_in[pb+1];
    const float* convw  = (const float*)d_in[pb+2];
    const float* convb  = (const float*)d_in[pb+3];
    const float* xproj  = (const float*)d_in[pb+4];
    const float* dtW    = (const float*)d_in[pb+5];
    const float* dtb    = (const float*)d_in[pb+6];
    const float* Alog   = (const float*)d_in[pb+7];
    const float* Dp     = (const float*)d_in[pb+8];
    const float* outproj= (const float*)d_in[pb+9];

    k_rmsnorm<<<(unsigned)(ROWS/8),256,0,stream>>>(h,normw,xn16);
    k_inproj <<<(unsigned)(ROWS/64),256,0,stream>>>(xn16,inproj,xi,resb);
    k_conv   <<<(unsigned)(ROWS*DI/64/256),256,0,stream>>>(xi,convw,convb,u,u16);
    k_xproj  <<<(unsigned)(ROWS/128),256,0,stream>>>(u16,xproj,xdbl);
    k_delta  <<<(unsigned)(ROWS*DI/256),256,0,stream>>>(xdbl,dtW,dtb,delta);
    k_scan   <<<(unsigned)(S_*BB*64/8),256,0,stream>>>(delta,u,xdbl,resb,Alog,Dp,y16);
    k_outproj<<<(unsigned)(ROWS/128),256,0,stream>>>(y16,outproj,h);
  }

  k_finalmean<<<S_*BB,256,0,stream>>>(h,fnorm,hm);
  k_head   <<<(S_*BB*EMB+255)/256,256,0,stream>>>(hm,outW,outb,out);
  k_combine<<<(BB*EMB+255)/256,256,0,stream>>>(out);
}
